// FFB_encoder_83305185673777
// MI455X (gfx1250) — matrix operand order repaired and checked by reading, not compiled, not run on hardware
//
#include <hip/hip_runtime.h>

typedef __attribute__((ext_vector_type(16))) _Float16 v16h;
typedef __attribute__((ext_vector_type(8)))  _Float16 v8h;
typedef __attribute__((ext_vector_type(8)))  float    v8f;
typedef __attribute__((ext_vector_type(4)))  int      v4i;

typedef __attribute__((address_space(1))) v4i gv4i;   // global int4
typedef __attribute__((address_space(3))) v4i lv4i;   // LDS int4

#define HID   128
#define LV    7
#define GFW   17          // grid_feats row width = 3 + 7*2
#define OUTW  131         // 3 + 128
#define TWO_PI 6.28318530717958647692f
#define KSIN  8.91267681314613790f    // 56 / (2*pi): hw v_sin computes sin(2*pi*x)
#define BLK   512         // 16 waves / block
#define ROWS  256         // rows per block (16 per wave)

#if __has_builtin(__builtin_amdgcn_global_load_async_to_lds_b128)
#define HAVE_ASYNC 1
#else
#define HAVE_ASYNC 0
#endif

// hardware sin: computes sin(2*pi*x)
__device__ __forceinline__ float hsin(float x) {
#if __has_builtin(__builtin_amdgcn_sinf)
    return __builtin_amdgcn_sinf(x);
#else
    return __sinf(TWO_PI * x);
#endif
}

// ---------------------------------------------------------------------------
// Prep: convert W_mid / W_high (fp32 [7][128][128]) into f16 WMMA B-fragments.
//   frag base (f16 units) = ((which*7 + l) * 16384) + ((j*4 + t)*32 + lane)*16
//   lane holds: col h = 16*j + (lane&15), K = 32*t + 16*(lane>>4) + {0..15}
// ---------------------------------------------------------------------------
__global__ __launch_bounds__(256) void ffb_prep_w(const float* __restrict__ Wmid,
                                                  const float* __restrict__ Whigh,
                                                  _Float16* __restrict__ wsW) {
    int tid = blockIdx.x * 256 + threadIdx.x;          // 0 .. 14335
    if (tid >= 2 * LV * 8 * 4 * 32) return;
    int lane  = tid & 31;
    int t     = (tid >> 5) & 3;
    int j     = (tid >> 7) & 7;
    int wl    = tid >> 10;                             // which*7 + l
    int l     = wl % LV;
    int which = wl / LV;
    const float* W = (which == 0 ? Wmid : Whigh) + l * HID * HID;
    int h  = j * 16 + (lane & 15);
    int kb = t * 32 + (lane >> 4) * 16;
    _Float16* dst = wsW + tid * 16;
    const float* src = W + h * HID + kb;
#pragma unroll
    for (int i = 0; i < 16; ++i) dst[i] = (_Float16)src[i];
}

// ---------------------------------------------------------------------------
// Stage one layer's mid+high fragment blocks (32 KB each) into LDS.
// ---------------------------------------------------------------------------
__device__ __forceinline__ void stage_layer(const _Float16* __restrict__ wsW, int l,
                                            _Float16* dstMid, _Float16* dstHigh,
                                            int tid) {
    const _Float16* gm = wsW + (size_t)(0 * LV + l) * 16384;
    const _Float16* gh = wsW + (size_t)(1 * LV + l) * 16384;
#if HAVE_ASYNC
    gv4i* gpm = (gv4i*)(uintptr_t)gm;
    gv4i* gph = (gv4i*)(uintptr_t)gh;
    lv4i* lpm = (lv4i*)(unsigned)(uintptr_t)dstMid;     // low 32 bits = LDS offset
    lv4i* lph = (lv4i*)(unsigned)(uintptr_t)dstHigh;
    for (int i = tid; i < 2048; i += BLK) {             // 2048 x 16B = 32 KB each
        __builtin_amdgcn_global_load_async_to_lds_b128(gpm + i, lpm + i, 0, 0);
        __builtin_amdgcn_global_load_async_to_lds_b128(gph + i, lph + i, 0, 0);
    }
#else
    for (int i = tid; i < 2048; i += BLK) {
        ((v8h*)dstMid)[i]  = ((const v8h*)gm)[i];
        ((v8h*)dstHigh)[i] = ((const v8h*)gh)[i];
    }
#endif
}

__device__ __forceinline__ void wait_async_done() {
#if HAVE_ASYNC
    asm volatile("s_wait_asynccnt 0" ::: "memory");
#endif
}

// ---------------------------------------------------------------------------
// Main: 512 threads = 16 waves; wave w handles rows blockIdx.x*256 + w*16 ..+15
// ---------------------------------------------------------------------------
__global__ __launch_bounds__(BLK) void ffb_main(const float* __restrict__ in_pos,
                                                const float* __restrict__ grid_feats,
                                                const float* __restrict__ ffn_A,
                                                const float* __restrict__ ffn_sigma,
                                                const float* __restrict__ W0,
                                                const float* __restrict__ b0,
                                                const float* __restrict__ b_mid,
                                                const float* __restrict__ b_high,
                                                const _Float16* __restrict__ wsW,
                                                float* __restrict__ out,
                                                int nrows) {
    __shared__ _Float16 wbuf[2][2][8 * 4 * 32 * 16]; // 128 KB double-buffered weights
    __shared__ _Float16 xtile[16][16 * HID];         // 64 KB per-wave activation tiles
    __shared__ float As[LV][2][HID];                 // sigma*A (2pi folded into hw sin)
    __shared__ float W0s[HID][3];                    // pre-scaled by KSIN
    __shared__ float b0s[HID];                       // pre-scaled by KSIN
    __shared__ float bmids[LV][HID];                 // pre-scaled by KSIN
    __shared__ float bhighs[LV][HID];                // pre-scaled by KSIN
    __shared__ float gxs[ROWS][14];
    __shared__ float poss[ROWS][3];

    const int tid  = threadIdx.x;
    const int w    = tid >> 5;
    const int lane = tid & 31;
    const int rowBase = blockIdx.x * ROWS;

    // kick off async staging of layer 0 weights immediately
    stage_layer(wsW, 0, wbuf[0][0], wbuf[0][1], tid);

    // ---- stage block-wide constants into LDS (scale factors folded) ------
    for (int i = tid; i < HID * 3; i += BLK) ((float*)W0s)[i] = W0[i] * KSIN;
    for (int i = tid; i < HID; i += BLK)     b0s[i] = b0[i] * KSIN;
    for (int i = tid; i < LV * HID; i += BLK) {
        ((float*)bmids)[i]  = b_mid[i]  * KSIN;
        ((float*)bhighs)[i] = b_high[i] * KSIN;
    }
    for (int i = tid; i < LV * 2 * HID; i += BLK)
        ((float*)As)[i] = ffn_A[i] * ffn_sigma[i >> 8];   // 2pi cancels with hw sin

    // ---- per-row staging + pos01 output ----------------------------------
    if (tid < ROWS) {
        int row = rowBase + tid;
        if (row < nrows) {
            const float* gf = grid_feats + (size_t)row * GFW + 3;
#pragma unroll
            for (int f = 0; f < 14; ++f) gxs[tid][f] = gf[f];
            float p0 = in_pos[row * 3 + 0];
            float p1 = in_pos[row * 3 + 1];
            float p2 = in_pos[row * 3 + 2];
            poss[tid][0] = p0; poss[tid][1] = p1; poss[tid][2] = p2;
            float* o = out + (size_t)row * OUTW;
            o[0] = (p0 + 1.0f) * 0.5f;
            o[1] = (p1 + 1.0f) * 0.5f;
            o[2] = (p2 + 1.0f) * 0.5f;
        }
    }
    __syncthreads();

    // ---- first SIREN layer: x0 = sin(56*(pos @ W0^T + b0)) ---------------
    {
        int m  = lane & 15;
        int hb = (lane >> 4) * 64;
        float p0 = poss[w * 16 + m][0];
        float p1 = poss[w * 16 + m][1];
        float p2 = poss[w * 16 + m][2];
        for (int h = hb; h < hb + 64; ++h) {
            float s = b0s[h] + W0s[h][0] * p0 + W0s[h][1] * p1 + W0s[h][2] * p2;
            xtile[w][m * HID + h] = (_Float16)hsin(s);
        }
    }

    // C/D layout indices for this lane
    const int moff = (lane >> 4) * 8;     // row offset for C/D elements
    const int hC   = lane & 15;           // col within 16-wide n-tile
    const int mA   = lane & 15;           // A-fragment row
    const int kAo  = (lane >> 4) * 16;    // A-fragment K sub-offset

    v8f buf[8];
#pragma unroll
    for (int j = 0; j < 8; ++j) buf[j] = (v8f){};

    // layer-0 weights must be resident before first use
    wait_async_done();
    __syncthreads();

    for (int l = 0; l < LV; ++l) {
        // prefetch next layer's weights into the other buffer (overlaps compute)
        if (l + 1 < LV)
            stage_layer(wsW, l + 1, wbuf[(l + 1) & 1][0], wbuf[(l + 1) & 1][1], tid);

        const _Float16* Bm = wbuf[l & 1][0];
        const _Float16* Bh = wbuf[l & 1][1];

        // ---- load A fragments (current x) from LDS -----------------------
        v16h a[4];
#pragma unroll
        for (int t = 0; t < 4; ++t) {
            const _Float16* src = &xtile[w][mA * HID + t * 32 + kAo];
            ((v8h*)&a[t])[0] = *(const v8h*)(src);
            ((v8h*)&a[t])[1] = *(const v8h*)(src + 8);
        }

        // ---- mid GEMM: y = sin(56*(x Wm^T + bm)); x = grid + y -----------
#pragma unroll
        for (int j = 0; j < 8; ++j) {
            v8f acc = (v8f){};
#pragma unroll
            for (int t = 0; t < 4; ++t) {
                v16h b;
                const _Float16* bp = Bm + ((j * 4 + t) * 32 + lane) * 16;
                ((v8h*)&b)[0] = *(const v8h*)(bp);
                ((v8h*)&b)[1] = *(const v8h*)(bp + 8);
                acc = __builtin_amdgcn_wmma_f32_16x16x32_f16(
                    false, a[t], false, b, (short)0, acc, false, false);
            }
            int h  = j * 16 + hC;
            float A0 = As[l][0][h], A1 = As[l][1][h];
            float bm = bmids[l][h];
#pragma unroll
            for (int v = 0; v < 8; ++v) {
                int m = v + moff;
                float g0 = gxs[w * 16 + m][2 * l + 0];
                float g1 = gxs[w * 16 + m][2 * l + 1];
                float g  = hsin(fmaf(g0, A0, g1 * A1));
                float xn = g + hsin(fmaf(KSIN, acc[v], bm));
                xtile[w][m * HID + h] = (_Float16)xn;
            }
        }

        // ---- reload A fragments (new x) ----------------------------------
#pragma unroll
        for (int t = 0; t < 4; ++t) {
            const _Float16* src = &xtile[w][mA * HID + t * 32 + kAo];
            ((v8h*)&a[t])[0] = *(const v8h*)(src);
            ((v8h*)&a[t])[1] = *(const v8h*)(src + 8);
        }

        // ---- high GEMM: buf += sin(56*(x Wh^T + bh)) ---------------------
#pragma unroll
        for (int j = 0; j < 8; ++j) {
            v8f acc = (v8f){};
#pragma unroll
            for (int t = 0; t < 4; ++t) {
                v16h b;
                const _Float16* bp = Bh + ((j * 4 + t) * 32 + lane) * 16;
                ((v8h*)&b)[0] = *(const v8h*)(bp);
                ((v8h*)&b)[1] = *(const v8h*)(bp + 8);
                acc = __builtin_amdgcn_wmma_f32_16x16x32_f16(
                    false, a[t], false, b, (short)0, acc, false, false);
            }
            float bh = bhighs[l][j * 16 + hC];
#pragma unroll
            for (int v = 0; v < 8; ++v)
                buf[j][v] += hsin(fmaf(KSIN, acc[v], bh));
        }

        // next-layer async copy must land + all waves done with this buffer
        wait_async_done();
        __syncthreads();
    }

    // ---- write buf/7 to out[:, 3:131] ------------------------------------
    const float inv7 = 1.0f / 7.0f;
#pragma unroll
    for (int j = 0; j < 8; ++j) {
#pragma unroll
        for (int v = 0; v < 8; ++v) {
            int m = v + moff;
            int h = j * 16 + hC;
            int row = rowBase + w * 16 + m;
            out[(size_t)row * OUTW + 3 + h] = buf[j][v] * inv7;
        }
    }
}

extern "C" void kernel_launch(void* const* d_in, const int* in_sizes, int n_in,
                              void* d_out, int out_size, void* d_ws, size_t ws_size,
                              hipStream_t stream) {
    const float* in_pos     = (const float*)d_in[0];
    const float* grid_feats = (const float*)d_in[1];
    const float* ffn_A      = (const float*)d_in[2];
    const float* ffn_sigma  = (const float*)d_in[3];
    const float* W0         = (const float*)d_in[4];
    const float* b0         = (const float*)d_in[5];
    const float* W_mid      = (const float*)d_in[6];
    const float* b_mid      = (const float*)d_in[7];
    const float* W_high     = (const float*)d_in[8];
    const float* b_high     = (const float*)d_in[9];
    float* out = (float*)d_out;
    _Float16* wsW = (_Float16*)d_ws;      // 448 KB of f16 B-fragments

    int nrows = in_sizes[0] / 3;          // N

    // 2*7*8*4*32 = 14336 fragment-lanes, 16 f16 each
    ffb_prep_w<<<14336 / 256, 256, 0, stream>>>(W_mid, W_high, wsW);

    int nblocks = (nrows + ROWS - 1) / ROWS;
    ffb_main<<<nblocks, BLK, 0, stream>>>(in_pos, grid_feats, ffn_A, ffn_sigma,
                                          W0, b0, b_mid, b_high, wsW, out, nrows);
}
